// Text2Motion_Transformer_86096914416096
// MI455X (gfx1250) — matrix operand order repaired and checked by reading, not compiled, not run on hardware
//
#include <hip/hip_runtime.h>
#include <cstddef>

// ---------------- constants (match reference) ----------------
#define LAYERS 9
#define Bc 32
#define Tc 256
#define Cc 512
#define Hc 8
#define HDc 64
#define FCc 2048
#define NREL 513        // 2*MAXREL+1
#define Mrows (Bc*Tc)   // 8192

typedef __attribute__((ext_vector_type(16))) __bf16 bf16x16;
typedef __attribute__((ext_vector_type(8)))  float  floatx8;
typedef unsigned int u32x4 __attribute__((ext_vector_type(4)));

// ---------------- bf16 helpers (raw u16 storage) ----------------
static __device__ __forceinline__ unsigned short f2bf(float f) {
    unsigned int x = __builtin_bit_cast(unsigned int, f);
    unsigned int lsb = (x >> 16) & 1u;
    x += 0x7fffu + lsb;                 // round-to-nearest-even
    return (unsigned short)(x >> 16);
}
static __device__ __forceinline__ float bf2f(unsigned short u) {
    unsigned int x = ((unsigned int)u) << 16;
    return __builtin_bit_cast(float, x);
}

// =================================================================
// Generic batched bf16 WMMA GEMM:  C[z] = A[z] (MxK) * Bt[z] (NxK)^T
//   Tile template: BM x BN block tile, per-wave AM*16 (M) x BNF*16 (N),
//   waves = (BM/(16*AM))*(BN/(16*BNF)), threads = BM*2 (= one 32B
//   staging slot per thread per tile).  K step 32, double-buffered LDS,
//   one barrier per K-step.
// Batch offset(z) = (z/div)*s1 + (z%div)*s2 for each operand.
// Epilogue: +bias[n], optional exact GELU, +residual, f32/bf16 stores.
// =================================================================
template<int BM, int BN, int AM, int BNF>
__global__ __launch_bounds__(BM * 2) void gemm_bf16_wmma(
    const unsigned short* __restrict__ A,  long long aS1, long long aS2, int aDiv, int lda,
    const unsigned short* __restrict__ Bt, long long bS1, long long bS2, int bDiv, int ldb,
    float* __restrict__ outF, unsigned short* __restrict__ outB,
    long long oS1, long long oS2, int oDiv, int ldo,
    const float* __restrict__ bias, const float* __restrict__ resid,
    int M, int N, int K, int gelu)
{
    static_assert(BM == BN, "staging assumes square block tile");
    constexpr int MW    = BM / (16 * AM);          // waves along M
    constexpr int WAVES = MW * (BN / (16 * BNF));  // total waves
    static_assert(WAVES * 32 == BM * 2, "threads must equal staging slots");
    constexpr int LST = 40;                        // LDS row stride (halves), 16B aligned + bank spread

    __shared__ __align__(16) unsigned short sA[2][BM * LST];
    __shared__ __align__(16) unsigned short sB[2][BN * LST];

    const int tid = threadIdx.x;
    const int z   = blockIdx.z;
    const long long aOff = (long long)(z / aDiv) * aS1 + (long long)(z % aDiv) * aS2;
    const long long bOff = (long long)(z / bDiv) * bS1 + (long long)(z % bDiv) * bS2;
    const long long oOff = (long long)(z / oDiv) * oS1 + (long long)(z % oDiv) * oS2;

    const int m0 = blockIdx.y * BM;
    const int n0 = blockIdx.x * BN;

    const int wid  = tid >> 5;
    const int lane = tid & 31;
    const int lh   = lane & 15;
    const bool lo  = lane < 16;
    const int wmo  = (wid % MW) * 16 * AM;   // wave M origin in tile
    const int wno  = (wid / MW) * 16 * BNF;  // wave N origin in tile

    const int srow = tid >> 1;               // staging row (0..BM-1)
    const int sseg = (tid & 1) * 16;         // 0 or 16 halves

    const unsigned short* gA = A + aOff + (long long)(m0 + srow) * lda + sseg;
    const unsigned short* gB = Bt + bOff + (long long)(n0 + srow) * ldb + sseg;

    auto stage = [&](int buf, int k0) {
        const u32x4* sa = (const u32x4*)(gA + k0);
        u32x4* da = (u32x4*)(sA[buf] + srow * LST + sseg);
        da[0] = sa[0]; da[1] = sa[1];
        const u32x4* sb = (const u32x4*)(gB + k0);
        u32x4* db = (u32x4*)(sB[buf] + srow * LST + sseg);
        db[0] = sb[0]; db[1] = sb[1];
    };

    floatx8 acc[AM][BNF] = {};

    const int nsteps = K / 32;
    stage(0, 0);
    __syncthreads();

    union Frag { u32x4 q[2]; bf16x16 v; };

    for (int s = 0; s < nsteps; ++s) {
        if (s + 1 < nsteps) stage((s + 1) & 1, (s + 1) * 32);

        const unsigned short* bufA = sA[s & 1];
        const unsigned short* bufB = sB[s & 1];

        // A fragments: 16x32, lanes 0-15 hold K{0..7,16..23}, lanes 16-31 K{8..15,24..31}
        Frag fa[AM];
        #pragma unroll
        for (int am = 0; am < AM; ++am) {
            const unsigned short* ap = bufA + (wmo + 16 * am + lh) * LST + (lo ? 0 : 8);
            fa[am].q[0] = *(const u32x4*)ap;
            fa[am].q[1] = *(const u32x4*)(ap + 16);
        }
        // B fragments: lane = N, lanes 0-15 hold K{0..7,16..23}, lanes 16-31 K{8..15,24..31}
        #pragma unroll
        for (int nt = 0; nt < BNF; ++nt) {
            Frag fb;
            const unsigned short* bp = bufB + (wno + nt * 16 + lh) * LST + (lo ? 0 : 8);
            fb.q[0] = *(const u32x4*)bp;
            fb.q[1] = *(const u32x4*)(bp + 16);
            #pragma unroll
            for (int am = 0; am < AM; ++am) {
                acc[am][nt] = __builtin_amdgcn_wmma_f32_16x16x32_bf16(
                    false, fa[am].v, false, fb.v, (short)0, acc[am][nt], false, false);
            }
        }
        __syncthreads();
    }

    // ---- epilogue: D layout lanes0-15 M=r,N=lane; lanes16-31 M=8+r,N=lane-16
    #pragma unroll
    for (int am = 0; am < AM; ++am) {
        #pragma unroll
        for (int nt = 0; nt < BNF; ++nt) {
            const int n = n0 + wno + nt * 16 + lh;
            const float bv = bias ? bias[n] : 0.0f;
            #pragma unroll
            for (int r = 0; r < 8; ++r) {
                const int m = m0 + wmo + 16 * am + r + (lo ? 0 : 8);
                float v = acc[am][nt][r] + bv;
                if (gelu) v = 0.5f * v * (1.0f + erff(v * 0.70710678118654752f));
                const long long idx = oOff + (long long)m * ldo + n;
                if (resid) v += resid[idx];
                if (outF) outF[idx] = v;
                if (outB) outB[idx] = f2bf(v);
            }
        }
    }
}

// =================================================================
// LayerNorm (f32 in) -> bf16 out.  One wave per row of C=512.
// =================================================================
__global__ __launch_bounds__(256) void layernorm_bf16(
    const float* __restrict__ x, const float* __restrict__ g,
    const float* __restrict__ b, unsigned short* __restrict__ out, int rows)
{
    const int wid = threadIdx.x >> 5, lane = threadIdx.x & 31;
    const int row = blockIdx.x * 8 + wid;
    if (row >= rows) return;
    const float* xr = x + (long long)row * Cc;
    float s = 0.f, ss = 0.f;
    #pragma unroll
    for (int j = 0; j < Cc / 32; ++j) { float v = xr[lane + 32 * j]; s += v; ss += v * v; }
    #pragma unroll
    for (int o = 16; o > 0; o >>= 1) { s += __shfl_xor(s, o, 32); ss += __shfl_xor(ss, o, 32); }
    const float m   = s / Cc;
    const float var = ss / Cc - m * m;
    const float inv = rsqrtf(var + 1e-5f);
    #pragma unroll
    for (int j = 0; j < Cc / 32; ++j) {
        const int e = lane + 32 * j;
        out[(long long)row * Cc + e] = f2bf((xr[e] - m) * inv * g[e] + b[e]);
    }
}

// =================================================================
// Softmax with fused rel-k bias + mask + scale.  One wave per (b,h,q) row.
// bias[q,k] = dot(q_row, rel_k[clip(q-k)+256])
// =================================================================
__global__ __launch_bounds__(256) void softmax_relk(
    const float* __restrict__ S, const float* __restrict__ qf,
    const unsigned char* __restrict__ mask, const float* __restrict__ relk,
    unsigned short* __restrict__ aout, float scale)
{
    __shared__ __align__(16) float sq[8][HDc];
    const int wid = threadIdx.x >> 5, lane = threadIdx.x & 31;
    const long long rw = (long long)blockIdx.x * 8 + wid;   // = (b*H+h)*T + q
    const int qi = (int)(rw % Tc);
    const int h  = (int)((rw / Tc) % Hc);
    const int b  = (int)(rw / ((long long)Hc * Tc));

    const float* qr = qf + ((long long)(b * Tc + qi)) * Cc + h * HDc;
    sq[wid][lane] = qr[lane];
    sq[wid][lane + 32] = qr[lane + 32];
    __syncthreads();

    const long long rowoff = rw * Tc;
    const float4* q4 = (const float4*)sq[wid];
    float vals[8];
    float mx = -3.4e38f;
    #pragma unroll
    for (int j = 0; j < 8; ++j) {
        const int k = lane + 32 * j;
        int d = qi - k; d = d < -256 ? -256 : (d > 256 ? 256 : d);
        const float4* rk = (const float4*)(relk + (long long)(d + 256) * HDc);
        float bias = 0.f;
        #pragma unroll
        for (int t = 0; t < HDc / 4; ++t) {
            float4 a = q4[t], c = rk[t];
            bias += a.x * c.x + a.y * c.y + a.z * c.z + a.w * c.w;
        }
        float v = (S[rowoff + k] + bias) * scale;
        if (!mask[rowoff + k]) v = -3.4e38f;
        vals[j] = v;
        mx = fmaxf(mx, v);
    }
    #pragma unroll
    for (int o = 16; o > 0; o >>= 1) mx = fmaxf(mx, __shfl_xor(mx, o, 32));
    float sum = 0.f;
    #pragma unroll
    for (int j = 0; j < 8; ++j) { vals[j] = expf(vals[j] - mx); sum += vals[j]; }
    #pragma unroll
    for (int o = 16; o > 0; o >>= 1) sum += __shfl_xor(sum, o, 32);
    const float inv = 1.0f / sum;
    #pragma unroll
    for (int j = 0; j < 8; ++j) aout[rowoff + lane + 32 * j] = f2bf(vals[j] * inv);
}

// =================================================================
// y += a @ rel_v[idx]   (VALU; a row staged in LDS), writes bf16 y.
// One block (64 threads = head dim) per (b,h,q) row.
// =================================================================
__global__ __launch_bounds__(64) void apply_relv(
    const float* __restrict__ yf, const unsigned short* __restrict__ a,
    const float* __restrict__ relv, unsigned short* __restrict__ ybf)
{
    __shared__ __align__(16) unsigned short sa[Tc];
    const long long rb = blockIdx.x;               // (b*H+h)*T + q
    const int d  = threadIdx.x;
    const int qi = (int)(rb % Tc);
    const int h  = (int)((rb / Tc) % Hc);
    const int b  = (int)(rb / ((long long)Hc * Tc));
    for (int k = d; k < Tc; k += 64) sa[k] = a[rb * Tc + k];
    __syncthreads();
    const long long yidx = ((long long)(b * Tc + qi)) * Cc + h * HDc + d;
    float v = yf[yidx];
    for (int k = 0; k < Tc; ++k) {
        int dd = qi - k; dd = dd < -256 ? -256 : (dd > 256 ? 256 : dd);
        v += bf2f(sa[k]) * relv[(long long)(dd + 256) * HDc + d];
    }
    ybf[yidx] = f2bf(v);
}

// =================================================================
// v_bf [B,T,C] -> vT [B,H,HD,T] (bf16 remap)
// =================================================================
__global__ __launch_bounds__(256) void vtrans(const unsigned short* __restrict__ v,
                                              unsigned short* __restrict__ vt)
{
    const long long i = (long long)blockIdx.x * 256 + threadIdx.x;
    if (i >= (long long)Bc * Tc * Cc) return;
    const int c = (int)(i % Cc);
    const long long bt = i / Cc;
    const int t = (int)(bt % Tc);
    const int b = (int)(bt / Tc);
    const int h = c / HDc, d = c % HDc;
    vt[((long long)((b * Hc + h) * HDc + d)) * Tc + t] = v[i];
}

// =================================================================
// W f32 [K,N] -> Wt bf16 [N,K]
// =================================================================
__global__ __launch_bounds__(256) void wtrans(const float* __restrict__ W,
                                              unsigned short* __restrict__ Wt,
                                              int Kd, int Nd)
{
    const long long i = (long long)blockIdx.x * 256 + threadIdx.x;
    if (i >= (long long)Kd * Nd) return;
    const int k = (int)(i % Kd);
    const long long n = i / Kd;
    Wt[i] = f2bf(W[(long long)k * Nd + n]);
}

// =================================================================
// Host orchestration
// =================================================================
extern "C" void kernel_launch(void* const* d_in, const int* in_sizes, int n_in,
                              void* d_out, int out_size, void* d_ws, size_t ws_size,
                              hipStream_t stream) {
    const float*         x_in = (const float*)d_in[0];
    const unsigned char* msk  = (const unsigned char*)d_in[1];
    const float* ln1g = (const float*)d_in[2];
    const float* ln1b = (const float*)d_in[3];
    const float* ln2g = (const float*)d_in[4];
    const float* ln2b = (const float*)d_in[5];
    const float* Wq = (const float*)d_in[6];  const float* bq = (const float*)d_in[7];
    const float* Wk = (const float*)d_in[8];  const float* bk = (const float*)d_in[9];
    const float* Wv = (const float*)d_in[10]; const float* bv = (const float*)d_in[11];
    const float* Wo = (const float*)d_in[12]; const float* bo = (const float*)d_in[13];
    const float* relk = (const float*)d_in[14];
    const float* relv = (const float*)d_in[15];
    const float* W1 = (const float*)d_in[16]; const float* b1 = (const float*)d_in[17];
    const float* W2 = (const float*)d_in[18]; const float* b2 = (const float*)d_in[19];

    // ---- carve workspace ----
    char* base = (char*)d_ws; size_t off = 0;
    auto carve = [&](size_t bytes) -> void* {
        size_t a = (off + 255) & ~(size_t)255; off = a + bytes; return base + a;
    };
    typedef unsigned short us;
    us* wtq = (us*)carve((size_t)Cc * Cc * 2);
    us* wtk = (us*)carve((size_t)Cc * Cc * 2);
    us* wtv = (us*)carve((size_t)Cc * Cc * 2);
    us* wto = (us*)carve((size_t)Cc * Cc * 2);
    us* wt1 = (us*)carve((size_t)Cc * FCc * 2);
    us* wt2 = (us*)carve((size_t)FCc * Cc * 2);
    us* h_bf  = (us*)carve((size_t)Mrows * Cc * 2);
    us* q_bf  = (us*)carve((size_t)Mrows * Cc * 2);
    us* k_bf  = (us*)carve((size_t)Mrows * Cc * 2);
    us* v_bf  = (us*)carve((size_t)Mrows * Cc * 2);
    us* vT    = (us*)carve((size_t)Mrows * Cc * 2);
    us* y_bf  = (us*)carve((size_t)Mrows * Cc * 2);
    us* h2_bf = (us*)carve((size_t)Mrows * Cc * 2);
    us* f_bf  = (us*)carve((size_t)Mrows * FCc * 2);
    us* a_bf  = (us*)carve((size_t)Bc * Hc * Tc * Tc * 2);
    float* q_f    = (float*)carve((size_t)Mrows * Cc * 4);
    float* scores = (float*)carve((size_t)Bc * Hc * Tc * Tc * 4);
    float* y_f    = (float*)carve((size_t)Mrows * Cc * 4);
    float* xm     = (float*)carve((size_t)Mrows * Cc * 4);
    float* xO0    = (float*)carve((size_t)Mrows * Cc * 4);
    float* xO1    = (float*)carve((size_t)Mrows * Cc * 4);
    float* xOut[2] = {xO0, xO1};

    const float scale = 0.04419417382415922f;  // 1/sqrt(C)=1/sqrt(512)
    const long long sTC = (long long)Tc * Cc;     // 131072
    const long long sTT = (long long)Tc * Tc;     // 65536
    const long long sHT = (long long)HDc * Tc;    // 16384

    const float* cur = x_in;
    for (int l = 0; l < LAYERS; ++l) {
        float* outx = (l == LAYERS - 1) ? (float*)d_out : xOut[l & 1];

        // ---- cast+transpose this layer's weights to bf16 [N,K] ----
        wtrans<<<(Cc * Cc + 255) / 256, 256, 0, stream>>>(Wq + (size_t)l * Cc * Cc, wtq, Cc, Cc);
        wtrans<<<(Cc * Cc + 255) / 256, 256, 0, stream>>>(Wk + (size_t)l * Cc * Cc, wtk, Cc, Cc);
        wtrans<<<(Cc * Cc + 255) / 256, 256, 0, stream>>>(Wv + (size_t)l * Cc * Cc, wtv, Cc, Cc);
        wtrans<<<(Cc * Cc + 255) / 256, 256, 0, stream>>>(Wo + (size_t)l * Cc * Cc, wto, Cc, Cc);
        wtrans<<<(Cc * FCc + 255) / 256, 256, 0, stream>>>(W1 + (size_t)l * Cc * FCc, wt1, Cc, FCc);
        wtrans<<<(Cc * FCc + 255) / 256, 256, 0, stream>>>(W2 + (size_t)l * FCc * Cc, wt2, FCc, Cc);

        // ---- LN1 ----
        layernorm_bf16<<<Mrows / 8, 256, 0, stream>>>(cur, ln1g + l * Cc, ln1b + l * Cc, h_bf, Mrows);

        // ---- Q/K/V projections (8192x512x512), 128x128 tiles ----
        gemm_bf16_wmma<128,128,2,4><<<dim3(Cc / 128, Mrows / 128, 1), 256, 0, stream>>>(
            h_bf, 0, 0, 1, Cc, wtq, 0, 0, 1, Cc,
            q_f, q_bf, 0, 0, 1, Cc, bq + l * Cc, nullptr, Mrows, Cc, Cc, 0);
        gemm_bf16_wmma<128,128,2,4><<<dim3(Cc / 128, Mrows / 128, 1), 256, 0, stream>>>(
            h_bf, 0, 0, 1, Cc, wtk, 0, 0, 1, Cc,
            nullptr, k_bf, 0, 0, 1, Cc, bk + l * Cc, nullptr, Mrows, Cc, Cc, 0);
        gemm_bf16_wmma<128,128,2,4><<<dim3(Cc / 128, Mrows / 128, 1), 256, 0, stream>>>(
            h_bf, 0, 0, 1, Cc, wtv, 0, 0, 1, Cc,
            nullptr, v_bf, 0, 0, 1, Cc, bv + l * Cc, nullptr, Mrows, Cc, Cc, 0);
        vtrans<<<(Mrows * Cc + 255) / 256, 256, 0, stream>>>(v_bf, vT);

        // ---- scores = q @ k^T per (b,h): batch=256, 256x256x64 ----
        gemm_bf16_wmma<128,128,2,4><<<dim3(Tc / 128, Tc / 128, Bc * Hc), 256, 0, stream>>>(
            q_bf, sTC, HDc, Hc, Cc, k_bf, sTC, HDc, Hc, Cc,
            scores, nullptr, sTT, 0, 1, Tc, nullptr, nullptr, Tc, Tc, HDc, 0);

        // ---- softmax with fused rel-k bias + mask + scale ----
        softmax_relk<<<(Bc * Hc * Tc) / 8, 256, 0, stream>>>(
            scores, q_f, msk, relk + (size_t)l * NREL * HDc, a_bf, scale);

        // ---- y = a @ v per (b,h): batch=256, 256x64x256 -> heads layout ----
        gemm_bf16_wmma<64,64,1,4><<<dim3(HDc / 64, Tc / 64, Bc * Hc), 128, 0, stream>>>(
            a_bf, sTT, 0, 1, Tc, vT, sHT, 0, 1, Tc,
            y_f, nullptr, sTC, HDc, Hc, Cc, nullptr, nullptr, Tc, HDc, Tc, 0);

        // ---- y += a @ rel_v[idx], cast to bf16 ----
        apply_relv<<<Bc * Hc * Tc, 64, 0, stream>>>(
            y_f, a_bf, relv + (size_t)l * NREL * HDc, y_bf);

        // ---- xm = cur + y @ Wo + bo ----
        gemm_bf16_wmma<128,128,2,4><<<dim3(Cc / 128, Mrows / 128, 1), 256, 0, stream>>>(
            y_bf, 0, 0, 1, Cc, wto, 0, 0, 1, Cc,
            xm, nullptr, 0, 0, 1, Cc, bo + l * Cc, cur, Mrows, Cc, Cc, 0);

        // ---- LN2 + FFN ----
        layernorm_bf16<<<Mrows / 8, 256, 0, stream>>>(xm, ln2g + l * Cc, ln2b + l * Cc, h2_bf, Mrows);
        gemm_bf16_wmma<128,128,2,4><<<dim3(FCc / 128, Mrows / 128, 1), 256, 0, stream>>>(
            h2_bf, 0, 0, 1, Cc, wt1, 0, 0, 1, Cc,
            nullptr, f_bf, 0, 0, 1, FCc, b1 + l * FCc, nullptr, Mrows, FCc, Cc, 1);
        gemm_bf16_wmma<128,128,2,4><<<dim3(Cc / 128, Mrows / 128, 1), 256, 0, stream>>>(
            f_bf, 0, 0, 1, FCc, wt2, 0, 0, 1, FCc,
            outx, nullptr, 0, 0, 1, Cc, b2 + l * Cc, xm, Mrows, Cc, FCc, 0);

        cur = outx;
    }
}
